// NTXentLoss_27315992002982
// MI455X (gfx1250) — matrix operand order repaired and checked by reading, not compiled, not run on hardware
//
#include <hip/hip_runtime.h>
#include <math.h>
#include <stdint.h>

// ---------------------------------------------------------------------------
// NT-Xent loss, fused for MI455X (gfx1250).
//   loss = mean_i [ log( sum_{j!=i} exp(2*dot(zn_i,zn_j)) ) - 2*dot(zn_i,zn_pos(i)) ]
//
// Precision split (autocast-style): the 68.7-GFLOP Gram matrix runs in bf16
// inputs with fp32 WMMA accumulation (V_WMMA_F32_16X16X32_BF16); norms,
// exp/log reduction and the positive-pair term stay fp32. Rows are unit-
// normalized so |sim|<=2: plain exp-sum is overflow-safe, and the diagonal
// contributes exactly exp(2), subtracted analytically.
//
// Data path: B tiles (16 KB) double-buffered via global_load_async_to_lds_b128
// (ASYNCcnt + s_wait_asynccnt + split barriers); 256-row A panel LDS-resident
// (266 KB, padded stride -> conflict-free b128 gathers). Bsh declared first so
// all B fragment loads use 16-bit ds immediates; K loop fully unrolled; four
// independent accumulator chains clear the bf16 WMMA->WMMA hazard window.
// ---------------------------------------------------------------------------

#define TWO_N   8192
#define NHALF   4096
#define DDIM    512
#define MTILE   256               // rows per block: 8 waves x 2 strips x 16
#define NSPLIT  16                // column-range splits (grid = 32*16 = 512)
#define TILES_PER_BLK (TWO_N / 16 / NSPLIT)   // 32 column tiles per block
#define LSTRIDE 520               // 512 + 8 bf16 pad: rows 4 banks apart

typedef __attribute__((ext_vector_type(16))) __bf16 v16bf;
typedef __attribute__((ext_vector_type(8)))  __bf16 v8bf;
typedef __attribute__((ext_vector_type(4)))  __bf16 v4bf;
typedef __attribute__((ext_vector_type(8)))  float  v8f;

#if __has_builtin(__builtin_amdgcn_s_wait_asynccnt)
#define WAIT_ASYNC(n) __builtin_amdgcn_s_wait_asynccnt(n)
#else
#define WAIT_ASYNC(n) asm volatile("s_wait_asynccnt %0" :: "n"(n) : "memory")
#endif

// Per-lane 16B async copy global -> LDS (ASYNCcnt-tracked, bypasses VGPRs).
__device__ __forceinline__ void async_cp16(uint32_t lds_off, const void* gptr) {
    asm volatile("global_load_async_to_lds_b128 %0, %1, off"
                 :: "v"(lds_off), "v"(gptr) : "memory");
}

__device__ __forceinline__ v16bf cat16(v8bf lo, v8bf hi) {
    return __builtin_shufflevector(lo, hi, 0,1,2,3,4,5,6,7,8,9,10,11,12,13,14,15);
}

// ---------------------------------------------------------------------------
__global__ __launch_bounds__(256) void ntx_zero_kernel(float* p, int n) {
    const int i = blockIdx.x * 256 + threadIdx.x;
    if (i < n) p[i] = 0.0f;
}

// One wave per row: fp32 norm; emit bf16 normalized row + fp32 norm.
__global__ __launch_bounds__(256) void ntx_normalize_kernel(const float* __restrict__ z,
                                                            __bf16* __restrict__ znb,
                                                            float* __restrict__ norms) {
    const int lane = threadIdx.x & 31;
    const int wave = threadIdx.x >> 5;
    const int row  = blockIdx.x * 8 + wave;

    const float4* src = (const float4*)(z + (size_t)row * DDIM);
    __bf16*       dst = znb + (size_t)row * DDIM;

    float4 v[4];
    float ss = 0.0f;
#pragma unroll
    for (int t = 0; t < 4; ++t) {
        v[t] = src[lane + 32 * t];
        ss = fmaf(v[t].x, v[t].x, ss); ss = fmaf(v[t].y, v[t].y, ss);
        ss = fmaf(v[t].z, v[t].z, ss); ss = fmaf(v[t].w, v[t].w, ss);
    }
#pragma unroll
    for (int m = 1; m < 32; m <<= 1) ss += __shfl_xor(ss, m);

    const float nrm = fmaxf(sqrtf(ss), 1e-12f);
    const float inv = 1.0f / nrm;
#pragma unroll
    for (int t = 0; t < 4; ++t) {
        v4bf o = { (__bf16)(v[t].x * inv), (__bf16)(v[t].y * inv),
                   (__bf16)(v[t].z * inv), (__bf16)(v[t].w * inv) };
        *(v4bf*)(dst + lane * 4 + 128 * t) = o;
    }
    if (lane == 0) norms[row] = nrm;
}

// ---------------------------------------------------------------------------
// Fused Gram + exp-sum. Block = 256 rows (8 waves x two 16-row strips),
// sweeps a 32-tile column range; B tiles double-buffered via async copies.
// S[i] += sum_j exp(2*dot_ij) over this block's column range (incl diagonal).
__global__ __launch_bounds__(256) void ntx_simexp_kernel(const __bf16* __restrict__ znb,
                                                         float* __restrict__ S) {
    // Bsh FIRST: keeps every B fragment address within the 16-bit ds immediate.
    __shared__ __bf16 Bsh[2][16 * LSTRIDE];      // 2 x 16,640 B B tiles
    __shared__ __bf16 Ash[MTILE * LSTRIDE];      // 266,240 B A panel

    const int tid  = threadIdx.x;
    const int lane = tid & 31;
    const int wave = tid >> 5;          // 0..7 -> rows [wave*32, wave*32+32)
    const int l15  = lane & 15;
    const int half = lane >> 4;

    const int mblk    = blockIdx.x >> 4;          // 0..31
    const int nsplit  = blockIdx.x & 15;          // 0..15
    const int rowBase = mblk * MTILE;
    const int ntBase  = nsplit * TILES_PER_BLK;

    // ---- prologue: async-stage the 256x512 bf16 A panel (64 b128 / thread) --
#pragma unroll 4
    for (int i = 0; i < 64; ++i) {
        const int f   = tid + 256 * i;            // 16B units, 64 per row
        const int row = f >> 6, u = f & 63;
        async_cp16((uint32_t)(uintptr_t)&Ash[row * LSTRIDE + u * 8],
                   znb + (size_t)(rowBase + row) * DDIM + u * 8);
    }
    // first B tile (full K): 4 b128 per thread
#pragma unroll
    for (int i = 0; i < 4; ++i) {
        const int f = tid + 256 * i;
        const int row = f >> 6, u = f & 63;
        async_cp16((uint32_t)(uintptr_t)&Bsh[0][row * LSTRIDE + u * 8],
                   znb + (size_t)(ntBase * 16 + row) * DDIM + u * 8);
    }

    // Fragment gather bases (bf16 elements).
    const __bf16* Ar0 = &Ash[(wave * 32      + l15) * LSTRIDE];  // strip 0 row
    const __bf16* Ar1 = &Ash[(wave * 32 + 16 + l15) * LSTRIDE];  // strip 1 row
    const int aoff = 8 * half;           // A: K = 8*half+[0,8) and 16+8*half+[0,8)
    const int boff = 8 * half;           // B: K = 8*half+[0,8) and 16+8*half+[0,8)

    float racc0[8], racc1[8];
#pragma unroll
    for (int r = 0; r < 8; ++r) { racc0[r] = 0.0f; racc1[r] = 0.0f; }

    for (int t = 0; t < TILES_PER_BLK; ++t) {
        const int buf = t & 1;
        if (t + 1 < TILES_PER_BLK) {
            const int nt1 = ntBase + t + 1;
            __bf16* b1 = &Bsh[(t + 1) & 1][0];
#pragma unroll
            for (int i = 0; i < 4; ++i) {
                const int f = tid + 256 * i;
                const int row = f >> 6, u = f & 63;
                async_cp16((uint32_t)(uintptr_t)&b1[row * LSTRIDE + u * 8],
                           znb + (size_t)(nt1 * 16 + row) * DDIM + u * 8);
            }
            WAIT_ASYNC(4);                        // tile t landed; t+1 in flight
        } else {
            WAIT_ASYNC(0);
        }
        __syncthreads();                          // all waves' portions visible

        const __bf16* Br = &Bsh[buf][l15 * LSTRIDE];

        // Four independent accumulator chains (strip x K-parity): each chain
        // re-issued only every 4th WMMA -> clears the bf16 hazard window.
        v8f c0a = {}, c1a = {}, c0b = {}, c1b = {};
#pragma unroll
        for (int ks = 0; ks < 16; ks += 2) {
            const int k0 = ks * 32;
            const int k1 = k0 + 32;
            v16bf bfA = cat16(*(const v8bf*)(Br  + k0 + boff),
                              *(const v8bf*)(Br  + k0 + boff + 16));
            v16bf a0A = cat16(*(const v8bf*)(Ar0 + k0 + aoff),
                              *(const v8bf*)(Ar0 + k0 + aoff + 16));
            v16bf a1A = cat16(*(const v8bf*)(Ar1 + k0 + aoff),
                              *(const v8bf*)(Ar1 + k0 + aoff + 16));
            v16bf bfB = cat16(*(const v8bf*)(Br  + k1 + boff),
                              *(const v8bf*)(Br  + k1 + boff + 16));
            v16bf a0B = cat16(*(const v8bf*)(Ar0 + k1 + aoff),
                              *(const v8bf*)(Ar0 + k1 + aoff + 16));
            v16bf a1B = cat16(*(const v8bf*)(Ar1 + k1 + aoff),
                              *(const v8bf*)(Ar1 + k1 + aoff + 16));
            c0a = __builtin_amdgcn_wmma_f32_16x16x32_bf16(
                      false, a0A, false, bfA, (short)0, c0a, false, false);
            c1a = __builtin_amdgcn_wmma_f32_16x16x32_bf16(
                      false, a1A, false, bfA, (short)0, c1a, false, false);
            c0b = __builtin_amdgcn_wmma_f32_16x16x32_bf16(
                      false, a0B, false, bfB, (short)0, c0b, false, false);
            c1b = __builtin_amdgcn_wmma_f32_16x16x32_bf16(
                      false, a1B, false, bfB, (short)0, c1b, false, false);
        }
        // Tile complete (full K): merge chains, exp-accumulate. sim=2*dot.
#pragma unroll
        for (int r = 0; r < 8; ++r) {
            racc0[r] += __expf(2.0f * (c0a[r] + c0b[r]));
            racc1[r] += __expf(2.0f * (c1a[r] + c1b[r]));
        }
        __syncthreads();                          // reads done before buffer reuse
    }

    // Rows are wave-exclusive: reduce across each 16-lane half, one atomic/row.
#pragma unroll
    for (int r = 0; r < 8; ++r) {
        float v0 = racc0[r], v1 = racc1[r];
        v0 += __shfl_xor(v0, 1); v1 += __shfl_xor(v1, 1);
        v0 += __shfl_xor(v0, 2); v1 += __shfl_xor(v1, 2);
        v0 += __shfl_xor(v0, 4); v1 += __shfl_xor(v1, 4);
        v0 += __shfl_xor(v0, 8); v1 += __shfl_xor(v1, 8);
        if (l15 == 0) {
            atomicAdd(&S[rowBase + wave * 32      + r + 8 * half], v0);
            atomicAdd(&S[rowBase + wave * 32 + 16 + r + 8 * half], v1);
        }
    }
}

// ---------------------------------------------------------------------------
// One wave per row: fp32 positive-pair term from the ORIGINAL input + norms.
__global__ __launch_bounds__(256) void ntx_rowloss_kernel(const float* __restrict__ z,
                                                          const float* __restrict__ norms,
                                                          const float* __restrict__ S,
                                                          float* accum) {
    const int lane = threadIdx.x & 31;
    const int wave = threadIdx.x >> 5;
    const int i    = blockIdx.x * 8 + wave;
    const int j    = (i < NHALF) ? (i + NHALF) : (i - NHALF);

    const float4* a = (const float4*)(z + (size_t)i * DDIM);
    const float4* b = (const float4*)(z + (size_t)j * DDIM);

    float dot = 0.0f;
#pragma unroll
    for (int t = 0; t < 4; ++t) {
        const float4 va = a[lane + 32 * t];
        const float4 vb = b[lane + 32 * t];
        dot = fmaf(va.x, vb.x, dot); dot = fmaf(va.y, vb.y, dot);
        dot = fmaf(va.z, vb.z, dot); dot = fmaf(va.w, vb.w, dot);
    }
#pragma unroll
    for (int m = 1; m < 32; m <<= 1) dot += __shfl_xor(dot, m);

    if (lane == 0) {
        const float e2     = 7.38905609893065f;   // exp(2): analytic diagonal
        const float possim = 2.0f * dot / (norms[i] * norms[j]);
        const float term   = __logf(S[i] - e2) - possim;
        atomicAdd(accum, term);
    }
}

__global__ __launch_bounds__(64) void ntx_finalize_kernel(const float* accum, float* out) {
    if (threadIdx.x == 0) out[0] = accum[0] * (1.0f / (float)TWO_N);
}

// ---------------------------------------------------------------------------
extern "C" void kernel_launch(void* const* d_in, const int* in_sizes, int n_in,
                              void* d_out, int out_size, void* d_ws, size_t ws_size,
                              hipStream_t stream) {
    const float* z = (const float*)d_in[0];       // (8192, 512) fp32
    float* out     = (float*)d_out;               // scalar fp32

    char* ws = (char*)d_ws;
    __bf16* znb  = (__bf16*)ws;                                   // 8 MB bf16 rows
    float*  norms= (float*)(ws + (size_t)TWO_N * DDIM * 2);       // 8192 norms
    float*  S    = norms + TWO_N;                                 // 8192 exp sums
    float*  accum= S + TWO_N;                                     // scalar

    ntx_zero_kernel<<<(TWO_N + 1 + 255) / 256, 256, 0, stream>>>(S, TWO_N + 1);
    ntx_normalize_kernel<<<TWO_N / 8, 256, 0, stream>>>(z, znb, norms);
    ntx_simexp_kernel<<<(TWO_N / MTILE) * NSPLIT, 256, 0, stream>>>(znb, S);
    ntx_rowloss_kernel<<<TWO_N / 8, 256, 0, stream>>>(z, norms, S, accum);
    ntx_finalize_kernel<<<1, 64, 0, stream>>>(accum, out);
}
